// NAT_33792802685789
// MI455X (gfx1250) — matrix operand order repaired and checked by reading, not compiled, not run on hardware
//
#include <hip/hip_runtime.h>
#include <hip/hip_bf16.h>
#include <math.h>

// ---------------------------------------------------------------------------
// CDNA5 (gfx1250) NAT forward. wave32. WMMA f32_16x16x32_f16 for all GEMMs.
// ---------------------------------------------------------------------------

typedef __attribute__((ext_vector_type(16))) _Float16 v16h;
typedef __attribute__((ext_vector_type(8)))  float    v8f;

#define WAVE 32

__device__ __forceinline__ float wave_sum(float v) {
#pragma unroll
  for (int o = 16; o > 0; o >>= 1) v += __shfl_xor(v, o, WAVE);
  return v;
}
__device__ __forceinline__ float wave_max(float v) {
#pragma unroll
  for (int o = 16; o > 0; o >>= 1) v = fmaxf(v, __shfl_xor(v, o, WAVE));
  return v;
}

// ---------------------------------------------------------------------------
// GEMM: out[M,N] = A[M,K] * W[N,K]^T + bias[N] (+ residual[M,N]) (opt GELU)
// Block: 256 threads = 8 waves; tile 128(M) x 64(N) x 32(K), LDS double-
// buffered (one barrier per K step). Wave (wm 0..3, wn 0..1) computes a
// 32x32 output tile = 4 WMMA accumulators. No divergent guards in the hot
// loop: N % 64 == 0 and K % 32 == 0 by construction; M handled by clamping
// the staged row (rows >= M produce garbage only in D rows the epilogue
// never stores).
// ---------------------------------------------------------------------------
__global__ __launch_bounds__(256) void gemm_wmma_kernel(
    const float* __restrict__ A, const float* __restrict__ Wt,
    const float* __restrict__ bias, const float* __restrict__ resid,
    float* __restrict__ out, int M, int N, int K, int gelu) {
  __shared__ _Float16 As[2][128][40];   // +8 pad: bank-conflict avoidance
  __shared__ _Float16 Bs[2][64][40];

  const int tid  = threadIdx.x;
  const int lane = tid & 31;
  const int wave = tid >> 5;
  const int wm   = wave & 3;        // 0..3 -> 32-row sub-tile
  const int wn   = wave >> 2;       // 0..1 -> 32-col sub-tile
  const int m0   = blockIdx.y * 128;
  const int n0   = blockIdx.x * 64;

  // staging decomposition (uniform, branch-free):
  //   A: 128 rows x 32 cols, 2 threads/row, 16 contiguous floats each
  //   B:  64 rows x 32 cols, 4 threads/row,  8 contiguous floats each
  const int arow_l = tid >> 1;
  const int acol   = (tid & 1) * 16;
  int arow_g = m0 + arow_l; if (arow_g >= M) arow_g = M - 1;   // clamp, no branch
  const float* Abase = A  + (size_t)arow_g * K + acol;
  const int brow_l = tid >> 2;
  const int bcol   = (tid & 3) * 8;
  const float* Bbase = Wt + (size_t)(n0 + brow_l) * K + bcol;

  float areg[16];
  float breg[8];

  auto g_load = [&](int k0) {
#pragma unroll
    for (int e = 0; e < 16; ++e) areg[e] = Abase[k0 + e];
#pragma unroll
    for (int e = 0; e < 8; ++e)  breg[e] = Bbase[k0 + e];
  };
  auto l_store = [&](int p) {
#pragma unroll
    for (int e = 0; e < 16; ++e) As[p][arow_l][acol + e] = (_Float16)areg[e];
#pragma unroll
    for (int e = 0; e < 8; ++e)  Bs[p][brow_l][bcol + e] = (_Float16)breg[e];
  };

  v8f c00 = {}, c01 = {}, c10 = {}, c11 = {};

  // prologue: stage K-panel 0 into buffer 0
  g_load(0);
  l_store(0);
  __syncthreads();

  int cur = 0;
  for (int k0 = 0; k0 < K; k0 += 32) {
    const bool has_next = (k0 + 32) < K;
    if (has_next) {
      g_load(k0 + 32);                       // in flight under the WMMAs
      if (k0 + 64 < K)                       // warm L2/GL1 two panels ahead
        __builtin_prefetch(Abase + k0 + 64, 0, 1);
    }

    // ---- fragments (ISA 7.12.2 16-bit A 16x32 / B 32x16 layouts) ----
    const int half = lane >> 4;
    const int mr   = wm * 32 + (lane & 15);
    const int nc   = wn * 32 + (lane & 15);
    v16h a0, a1, b0, b1;
#pragma unroll
    for (int e = 0; e < 8; ++e) {
      a0[e]     = As[cur][mr][half * 8 + e];
      a0[8 + e] = As[cur][mr][16 + half * 8 + e];
      a1[e]     = As[cur][mr + 16][half * 8 + e];
      a1[8 + e] = As[cur][mr + 16][16 + half * 8 + e];
    }
#pragma unroll
    for (int e = 0; e < 16; ++e) {
      b0[e] = Bs[cur][nc][(e < 8) ? (half * 8 + e) : (16 + half * 8 + (e - 8))];
      b1[e] = Bs[cur][nc + 16][(e < 8) ? (half * 8 + e) : (16 + half * 8 + (e - 8))];
    }

    c00 = __builtin_amdgcn_wmma_f32_16x16x32_f16(false, a0, false, b0,
                                                 (short)0, c00, false, false);
    c01 = __builtin_amdgcn_wmma_f32_16x16x32_f16(false, a0, false, b1,
                                                 (short)0, c01, false, false);
    c10 = __builtin_amdgcn_wmma_f32_16x16x32_f16(false, a1, false, b0,
                                                 (short)0, c10, false, false);
    c11 = __builtin_amdgcn_wmma_f32_16x16x32_f16(false, a1, false, b1,
                                                 (short)0, c11, false, false);

    if (has_next) {
      l_store(cur ^ 1);                      // other bank: no barrier needed
      __syncthreads();                       // single barrier per K step
      cur ^= 1;
    }
  }

  // ---- epilogue: C/D layout = VGPR r: M=r (lanes 0-15) / M=r+8 (16-31) ----
  const int nlo = lane & 15;
  const int hi  = lane >> 4;
  const v8f* acc[2][2] = {{&c00, &c01}, {&c10, &c11}};
#pragma unroll
  for (int ti = 0; ti < 2; ++ti) {
#pragma unroll
    for (int tj = 0; tj < 2; ++tj) {
      const v8f cc = *acc[ti][tj];
      const int n = n0 + wn * 32 + tj * 16 + nlo;
      const float bn = bias[n];
#pragma unroll
      for (int r = 0; r < 8; ++r) {
        const int m = m0 + wm * 32 + ti * 16 + r + hi * 8;
        if (m >= M) continue;
        float v = cc[r] + bn;
        if (resid) v += resid[(size_t)m * N + n];
        if (gelu)  v = 0.5f * v * (1.0f + erff(v * 0.70710678118654752f));
        out[(size_t)m * N + n] = v;
      }
    }
  }
}

// ---------------------------------------------------------------------------
// Neighborhood attention (KS=7, d=32 == wave32). One wave per (b,h,i,j).
// qkv: [T, 3C] with feature order [s][head][d]. out: [T, C] (head-major).
// Block = 128 threads = 4 waves. Grid count divisible by 4 for all stages.
// ---------------------------------------------------------------------------
__global__ __launch_bounds__(128) void na2d_kernel(
    const float* __restrict__ qkv, const float* __restrict__ rpb,
    float* __restrict__ out, int H, int W, int heads) {
  const int d = 32;
  const int C = heads * d;
  __shared__ float logits[4][64];

  const int lane = threadIdx.x & 31;
  const int wv   = threadIdx.x >> 5;
  int wid = blockIdx.x * 4 + wv;
  const int j = wid % W; wid /= W;
  const int i = wid % H; wid /= H;
  const int h = wid % heads;
  const int b = wid / heads;

  const int t = (b * H + i) * W + j;
  const float scale = 0.17677669529663687f;  // 1/sqrt(32)
  const float qc = qkv[(size_t)t * 3 * C + h * d + lane] * scale;

  int si = i - 3; if (si < 0) si = 0; if (si > H - 7) si = H - 7;
  int sj = j - 3; if (sj < 0) sj = 0; if (sj > W - 7) sj = W - 7;
  const float* rp = rpb + (size_t)h * 13 * 13;

  for (int n = 0; n < 49; ++n) {
    const int di = n / 7, dj = n % 7;
    const int ii = si + di, jj = sj + dj;
    const int tn = (b * H + ii) * W + jj;
    const float kc = qkv[(size_t)tn * 3 * C + C + h * d + lane];
    const float dot = wave_sum(qc * kc);
    if (lane == 0)
      logits[wv][n] = dot + rp[(ii - i + 6) * 13 + (jj - j + 6)];
  }
  __syncthreads();

  // softmax over 49 entries, 32 lanes (lane handles n and n+32)
  const float l0 = logits[wv][lane];
  const float l1 = (lane + 32 < 49) ? logits[wv][lane + 32] : -3.0e38f;
  const float mx = wave_max(fmaxf(l0, l1));
  const float e0 = __expf(l0 - mx);
  const float e1 = (lane + 32 < 49) ? __expf(l1 - mx) : 0.0f;
  const float inv = 1.0f / wave_sum(e0 + e1);
  logits[wv][lane] = e0 * inv;
  if (lane + 32 < 49) logits[wv][lane + 32] = e1 * inv;
  __syncthreads();

  float acc = 0.0f;
  for (int n = 0; n < 49; ++n) {
    const int tn = (b * H + si + n / 7) * W + sj + n % 7;
    acc += logits[wv][n] * qkv[(size_t)tn * 3 * C + 2 * C + h * d + lane];
  }
  out[(size_t)t * C + h * d + lane] = acc;
}

// ---------------------------------------------------------------------------
// LayerNorm over last dim. One wave per row; 8 rows per 256-thread block.
// ---------------------------------------------------------------------------
__global__ __launch_bounds__(256) void ln_kernel(
    const float* __restrict__ in, float* __restrict__ out,
    const float* __restrict__ g, const float* __restrict__ bb, int T, int C) {
  const int lane = threadIdx.x & 31;
  const int row  = blockIdx.x * 8 + (threadIdx.x >> 5);
  if (row >= T) return;
  const float* p = in + (size_t)row * C;
  float s = 0.f, s2 = 0.f;
  for (int c = lane; c < C; c += 32) { const float v = p[c]; s += v; s2 += v * v; }
  s = wave_sum(s); s2 = wave_sum(s2);
  const float mean = s / C;
  const float rstd = rsqrtf(s2 / C - mean * mean + 1e-5f);
  float* o = out + (size_t)row * C;
  for (int c = lane; c < C; c += 32)
    o[c] = (p[c] - mean) * rstd * g[c] + bb[c];
}

// LayerNorm + NHWC->NCHW scatter (for the 4 stage outputs)
__global__ __launch_bounds__(256) void ln_nchw_kernel(
    const float* __restrict__ in, float* __restrict__ out,
    const float* __restrict__ g, const float* __restrict__ bb,
    int T, int C, int HW) {
  const int lane = threadIdx.x & 31;
  const int row  = blockIdx.x * 8 + (threadIdx.x >> 5);
  if (row >= T) return;
  const float* p = in + (size_t)row * C;
  float s = 0.f, s2 = 0.f;
  for (int c = lane; c < C; c += 32) { const float v = p[c]; s += v; s2 += v * v; }
  s = wave_sum(s); s2 = wave_sum(s2);
  const float mean = s / C;
  const float rstd = rsqrtf(s2 / C - mean * mean + 1e-5f);
  const int b = row / HW, hw = row % HW;
  float* o = out + (size_t)b * C * HW + hw;
  for (int c = lane; c < C; c += 32)
    o[(size_t)c * HW] = (p[c] - mean) * rstd * g[c] + bb[c];
}

// ---------------------------------------------------------------------------
// Direct 3x3 stride-2 pad-1 conv. Output NHWC. Input NCHW (flag) or NHWC.
// ---------------------------------------------------------------------------
__global__ __launch_bounds__(256) void conv3x3s2_kernel(
    const float* __restrict__ in, const float* __restrict__ w,
    const float* __restrict__ bias, float* __restrict__ out,
    int Bn, int Hin, int Win, int Cin, int Hout, int Wout, int Cout,
    int in_nchw) {
  const int id = blockIdx.x * blockDim.x + threadIdx.x;
  const int total = Bn * Hout * Wout * Cout;
  if (id >= total) return;
  const int oc = id % Cout; int r = id / Cout;
  const int ow = r % Wout; r /= Wout;
  const int oh = r % Hout;
  const int b  = r / Hout;

  float acc = bias ? bias[oc] : 0.0f;
  for (int kh = 0; kh < 3; ++kh) {
    const int ih = oh * 2 + kh - 1;
    if (ih < 0 || ih >= Hin) continue;
    for (int kw = 0; kw < 3; ++kw) {
      const int iw = ow * 2 + kw - 1;
      if (iw < 0 || iw >= Win) continue;
      const float* wp = w + ((size_t)oc * Cin * 9) + kh * 3 + kw;
      if (in_nchw) {
        const float* ip = in + ((size_t)b * Cin * Hin + ih) * Win + iw;
        for (int ic = 0; ic < Cin; ++ic)
          acc += ip[(size_t)ic * Hin * Win] * wp[(size_t)ic * 9];
      } else {
        const float* ip = in + (((size_t)b * Hin + ih) * Win + iw) * Cin;
        for (int ic = 0; ic < Cin; ++ic)
          acc += ip[ic] * wp[(size_t)ic * 9];
      }
    }
  }
  out[(((size_t)b * Hout + oh) * Wout + ow) * Cout + oc] = acc;
}

// ---------------------------------------------------------------------------
// Host orchestration
// ---------------------------------------------------------------------------
struct NatBlk {
  const float *fc1_b, *fc1_w, *fc2_b, *fc2_w;
  const float *n1_b, *n1_g, *n2_b, *n2_g;
  const float *proj_b, *proj_w, *qkv_b, *qkv_w, *rpb;
};

static inline void run_gemm(const float* A, const float* W, const float* bias,
                            const float* resid, float* out, int M, int N, int K,
                            int gelu, hipStream_t s) {
  dim3 grid(N / 64, (M + 127) / 128);
  gemm_wmma_kernel<<<grid, 256, 0, s>>>(A, W, bias, resid, out, M, N, K, gelu);
}

extern "C" void kernel_launch(void* const* d_in, const int* in_sizes, int n_in,
                              void* d_out, int out_size, void* d_ws,
                              size_t ws_size, hipStream_t stream) {
  (void)in_sizes; (void)n_in; (void)out_size; (void)ws_size;
  static const int DEP[4] = {3, 4, 6, 5};

  int idx = 0;
  const float* x = (const float*)d_in[idx++];

  // params pytree flatten order: 'levels' -> 'out_norms' -> 'tok'
  NatBlk blks[4][6];
  const float *dn_ln_b[3], *dn_ln_g[3], *dn_w[3];
  for (int i = 0; i < 4; ++i) {
    for (int d = 0; d < DEP[i]; ++d) {
      NatBlk& b = blks[i][d];                 // alphabetical block keys:
      b.fc1_b  = (const float*)d_in[idx++];
      b.fc1_w  = (const float*)d_in[idx++];
      b.fc2_b  = (const float*)d_in[idx++];
      b.fc2_w  = (const float*)d_in[idx++];
      b.n1_b   = (const float*)d_in[idx++];
      b.n1_g   = (const float*)d_in[idx++];
      b.n2_b   = (const float*)d_in[idx++];
      b.n2_g   = (const float*)d_in[idx++];
      b.proj_b = (const float*)d_in[idx++];
      b.proj_w = (const float*)d_in[idx++];
      b.qkv_b  = (const float*)d_in[idx++];
      b.qkv_w  = (const float*)d_in[idx++];
      b.rpb    = (const float*)d_in[idx++];
    }
    if (i < 3) {                              // 'down': ln_b < ln_g < w
      dn_ln_b[i] = (const float*)d_in[idx++];
      dn_ln_g[i] = (const float*)d_in[idx++];
      dn_w[i]    = (const float*)d_in[idx++];
    }
  }
  const float *on_b[4], *on_g[4];
  for (int i = 0; i < 4; ++i) {               // out_norms: 'b' < 'g'
    on_b[i] = (const float*)d_in[idx++];
    on_g[i] = (const float*)d_in[idx++];
  }
  const float* tok_b0   = (const float*)d_in[idx++];   // tok: b0,b1,ln_b,ln_g,w0,w1
  const float* tok_b1   = (const float*)d_in[idx++];
  const float* tok_ln_b = (const float*)d_in[idx++];
  const float* tok_ln_g = (const float*)d_in[idx++];
  const float* tok_w0   = (const float*)d_in[idx++];
  const float* tok_w1   = (const float*)d_in[idx++];

  // scratch layout (floats): bufA(qkv/stem) | bufH(hidden) | y | z
  float* ws   = (float*)d_ws;
  float* bufA = ws;                 // 1204224 floats (stem 802816 fits here too)
  float* bufH = bufA + 1204224;     // 1204224
  float* y    = bufH + 1204224;     // 401408
  float* z    = y    + 401408;      // 401408  (ln out / attn out / down out)
  float* outp = (float*)d_out;

  // ---- stem: conv(3->32,s2) NCHW-in, conv(32->64,s2), tok layernorm ----
  {
    int n0 = 2 * 112 * 112 * 32;
    conv3x3s2_kernel<<<(n0 + 255) / 256, 256, 0, stream>>>(
        x, tok_w0, tok_b0, bufA, 2, 224, 224, 3, 112, 112, 32, 1);
    int n1 = 2 * 56 * 56 * 64;
    conv3x3s2_kernel<<<(n1 + 255) / 256, 256, 0, stream>>>(
        bufA, tok_w1, tok_b1, y, 2, 112, 112, 32, 56, 56, 64, 0);
    int T = 2 * 56 * 56;
    ln_kernel<<<(T + 7) / 8, 256, 0, stream>>>(y, y, tok_ln_g, tok_ln_b, T, 64);
  }

  size_t out_off = 0;
  int H = 56;
  for (int i = 0; i < 4; ++i) {
    const int C = 64 << i;
    const int heads = 2 << i;
    const int T = 2 * H * H;
    const int hidC = 3 * C;

    for (int d = 0; d < DEP[i]; ++d) {
      const NatBlk& bk = blks[i][d];
      // attention branch
      ln_kernel<<<(T + 7) / 8, 256, 0, stream>>>(y, z, bk.n1_g, bk.n1_b, T, C);
      run_gemm(z, bk.qkv_w, bk.qkv_b, nullptr, bufA, T, 3 * C, C, 0, stream);
      {
        int waves = 2 * heads * H * H;              // divisible by 4
        na2d_kernel<<<waves / 4, 128, 0, stream>>>(bufA, bk.rpb, z, H, H, heads);
      }
      run_gemm(z, bk.proj_w, bk.proj_b, y, y, T, C, C, 0, stream);   // +resid
      // MLP branch
      ln_kernel<<<(T + 7) / 8, 256, 0, stream>>>(y, z, bk.n2_g, bk.n2_b, T, C);
      run_gemm(z, bk.fc1_w, bk.fc1_b, nullptr, bufH, T, hidC, C, 1, stream); // GELU
      run_gemm(bufH, bk.fc2_w, bk.fc2_b, y, y, T, C, hidC, 0, stream); // +resid
    }

    // stage output: layernorm -> NCHW segment of d_out
    ln_nchw_kernel<<<(T + 7) / 8, 256, 0, stream>>>(
        y, outp + out_off, on_g[i], on_b[i], T, C, H * H);
    out_off += (size_t)2 * C * H * H;

    // downsample (reads y before overwrite; conv -> z, ln -> y)
    if (i < 3) {
      const int Ho = H / 2;
      const int n = 2 * Ho * Ho * 2 * C;
      conv3x3s2_kernel<<<(n + 255) / 256, 256, 0, stream>>>(
          y, dn_w[i], nullptr, z, 2, H, H, C, Ho, Ho, 2 * C, 0);
      const int To = 2 * Ho * Ho;
      ln_kernel<<<(To + 7) / 8, 256, 0, stream>>>(z, y, dn_ln_g[i], dn_ln_b[i],
                                                  To, 2 * C);
      H = Ho;
    }
  }
}
